// EdgeBlock_36593121362170
// MI455X (gfx1250) — matrix operand order repaired and checked by reading, not compiled, not run on hardware
//
#include <hip/hip_runtime.h>
#include <hip/hip_bf16.h>

// ---------------------------------------------------------------------------
// EdgeBlock: 4-layer MLP (512->256->256->256->128) per edge + segment mean.
// f16 WMMA (v_wmma_f32_16x16x32_f16), f32 accumulate. K-loop is software-
// pipelined with double-buffered A/B fragments so B loads overlap WMMAs.
// ---------------------------------------------------------------------------

typedef __attribute__((ext_vector_type(16))) _Float16 v16h;
typedef __attribute__((ext_vector_type(8)))  _Float16 v8h;
typedef __attribute__((ext_vector_type(8)))  float    v8f;

union AFrag { v16h v; v8h h8[2]; };

#define FEAT   128
#define DIN    512
#define ROWPAD 8     // LDS row padding (halfs) -> A-frag loads hit distinct banks

// -------------------------------- weight pack ------------------------------
// Packed layout: tile = kk*(N/16)+wn ; out[(tile*32 + lane)*16 + h]
//   lane: n = wn*16 + (lane&15), K-range base = kk*32 + (lane>>4)*8
//   h   : k = base + (h>>3)*16 + (h&7)   (B 32x16 f16 fragment, ISA 7.12.2)
__global__ void pack_w_kernel(const float* __restrict__ W, _Float16* __restrict__ out,
                              int K, int Ncols) {
  int idx = blockIdx.x * blockDim.x + threadIdx.x;
  int total = K * Ncols;
  if (idx >= total) return;
  int h    = idx & 15;
  int lane = (idx >> 4) & 31;
  int tile = idx >> 9;
  int ntiles = Ncols >> 4;
  int wn = tile % ntiles;
  int kk = tile / ntiles;
  int n = wn * 16 + (lane & 15);
  int k = kk * 32 + (h >> 3) * 16 + (lane >> 4) * 8 + (h & 7);
  out[idx] = (_Float16)W[k * Ncols + n];
}

__global__ void zero_kernel(float* __restrict__ p, int n) {
  int i = blockIdx.x * blockDim.x + threadIdx.x;
  if (i < n) p[i] = 0.0f;
}

__global__ void count_kernel(const int* __restrict__ segid, float* __restrict__ counts, int N) {
  int e = blockIdx.x * blockDim.x + threadIdx.x;
  if (e < N) atomicAdd(counts + segid[e], 1.0f);
}

__global__ void finalize_kernel(const float* __restrict__ sums,
                                const float* __restrict__ counts,
                                const int* __restrict__ segid,
                                float* __restrict__ out, int N) {
  int idx = blockIdx.x * blockDim.x + threadIdx.x;
  if (idx >= N * FEAT) return;
  int e = idx >> 7;
  int c = idx & (FEAT - 1);
  int s = segid[e];
  out[idx] = sums[s * FEAT + c] / counts[s];
}

// ----------------------- pipelined GEMM pass (one wave) --------------------
// acc[TP] += A[16,K] @ W[:, (p..p+TP)*16]   (A f16 in LDS, W packed in global)
// Double-buffered modulo-2 schedule: loads for stage k+1 issue before the
// WMMAs of stage k retire, so global_load_b128 latency overlaps matrix ops.
template <int K, int NTILES, int TP>
__device__ __forceinline__ void gemm_pass(const _Float16* __restrict__ wpk,
                                          const _Float16* inB, int p,
                                          int m, int kh, int lane,
                                          v8f (&acc)[TP]) {
  constexpr int KS = K + ROWPAD;
  constexpr int KT = K / 32;          // 32-K steps (even for all layers)
  const _Float16* arow  = inB + m * KS + kh * 8;
  const _Float16* bbase = wpk + ((size_t)p * 32 + lane) * 16;

  AFrag af0, af1;
  v16h  bf0[TP], bf1[TP];

  // prologue: stage kk=0 -> buffer 0
  af0.h8[0] = *(const v8h*)(arow);
  af0.h8[1] = *(const v8h*)(arow + 16);
#pragma unroll
  for (int t = 0; t < TP; ++t) bf0[t] = *(const v16h*)(bbase + t * 512);

#pragma unroll 1
  for (int kk = 0; kk < KT; kk += 2) {
    // prefetch stage kk+1 -> buffer 1
    {
      const _Float16* a1 = arow + (kk + 1) * 32;
      af1.h8[0] = *(const v8h*)(a1);
      af1.h8[1] = *(const v8h*)(a1 + 16);
      const _Float16* b1 = bbase + (size_t)(kk + 1) * NTILES * 512;
#pragma unroll
      for (int t = 0; t < TP; ++t) bf1[t] = *(const v16h*)(b1 + t * 512);
    }
    // compute stage kk (buffer 0)
#pragma unroll
    for (int t = 0; t < TP; ++t)
      acc[t] = __builtin_amdgcn_wmma_f32_16x16x32_f16(
          false, af0.v, false, bf0[t], (short)0, acc[t], false, false);
    // prefetch stage kk+2 -> buffer 0 (clamped redundant load on last iter)
    {
      const int k2 = (kk + 2 < KT) ? (kk + 2) : kk;
      const _Float16* a2 = arow + k2 * 32;
      af0.h8[0] = *(const v8h*)(a2);
      af0.h8[1] = *(const v8h*)(a2 + 16);
      const _Float16* b2 = bbase + (size_t)k2 * NTILES * 512;
#pragma unroll
      for (int t = 0; t < TP; ++t) bf0[t] = *(const v16h*)(b2 + t * 512);
    }
    // compute stage kk+1 (buffer 1)
#pragma unroll
    for (int t = 0; t < TP; ++t)
      acc[t] = __builtin_amdgcn_wmma_f32_16x16x32_f16(
          false, af1.v, false, bf1[t], (short)0, acc[t], false, false);
  }
}

// ------------------------------- MLP layer ---------------------------------
template <int K, int NL>
__device__ __forceinline__ void mlp_layer(const _Float16* __restrict__ wpk,
                                          const float* __restrict__ bias,
                                          const _Float16* inB, _Float16* outB,
                                          int lane) {
  const int m  = lane & 15;   // A row / D column within tile
  const int kh = lane >> 4;   // K half-group select
  constexpr int NS = NL + ROWPAD;
  constexpr int NTILES = NL / 16;
#pragma unroll 1
  for (int p = 0; p < NTILES; p += 4) {
    v8f acc[4] = {};
    gemm_pass<K, NTILES, 4>(wpk, inB, p, m, kh, lane, acc);
    // epilogue: bias + relu, f32 -> f16 to LDS (C/D: n=lane&15, row=r+kh*8)
#pragma unroll
    for (int t = 0; t < 4; ++t) {
      const int ncol = (p + t) * 16 + m;
      const float bv = bias[ncol];
#pragma unroll
      for (int r = 0; r < 8; ++r) {
        float v = acc[t][r] + bv;
        v = fmaxf(v, 0.0f);
        outB[(r + kh * 8) * NS + ncol] = (_Float16)v;
      }
    }
  }
}

// ------------------------------- main kernel -------------------------------
// 256 threads = 8 waves; 128 edges per block; each wave owns 16 rows and a
// private LDS slab (no inter-wave barriers, EXEC stays all-ones for WMMA).
__global__ __launch_bounds__(256) void edgeblock_mlp_wmma(
    const float* __restrict__ g, const float* __restrict__ ns,
    const float* __restrict__ nr, const float* __restrict__ ef,
    const int* __restrict__ segid,
    const _Float16* __restrict__ wpk1, const _Float16* __restrict__ wpk2,
    const _Float16* __restrict__ wpk3, const _Float16* __restrict__ wpk4,
    const float* __restrict__ b1, const float* __restrict__ b2,
    const float* __restrict__ b3, const float* __restrict__ b4,
    float* __restrict__ sums, int N) {
  extern __shared__ _Float16 smem[];
  const int lane   = threadIdx.x & 31;
  const int waveId = threadIdx.x >> 5;
  // per-wave slab: bufA = 16 x (512+8) halfs (X, reused for H2),
  //                bufB = 16 x (256+8) halfs (H1, H3)
  _Float16* bufA = smem + waveId * (16 * (DIN + ROWPAD) + 16 * (256 + ROWPAD));
  _Float16* bufB = bufA + 16 * (DIN + ROWPAD);
  const int edgeBase = blockIdx.x * 128 + waveId * 16;

  // ---- stage X = concat(g, ns, nr, e) -> f16 LDS tile, zero-fill OOB rows ----
  {
    const int sel = lane >> 3;                 // 8 lanes per 128-col source
    const float* S = (sel == 0) ? g : (sel == 1) ? ns : (sel == 2) ? nr : ef;
    const int cbase = (lane & 7) * 16;
#pragma unroll 1
    for (int r = 0; r < 16; ++r) {
      const int edge = edgeBase + r;
      v16h hv = {};
      if (edge < N) {
        const float4* p4 = (const float4*)(S + (size_t)edge * FEAT + cbase);
#pragma unroll
        for (int i = 0; i < 4; ++i) {
          float4 f = p4[i];
          hv[4 * i + 0] = (_Float16)f.x;
          hv[4 * i + 1] = (_Float16)f.y;
          hv[4 * i + 2] = (_Float16)f.z;
          hv[4 * i + 3] = (_Float16)f.w;
        }
      }
      *(v16h*)(bufA + r * (DIN + ROWPAD) + lane * 16) = hv;
    }
  }

  mlp_layer<512, 256>(wpk1, b1, bufA, bufB, lane);   // X  -> H1
  mlp_layer<256, 256>(wpk2, b2, bufB, bufA, lane);   // H1 -> H2
  mlp_layer<256, 256>(wpk3, b3, bufA, bufB, lane);   // H2 -> H3

  // ---- layer 4: 256 -> 128, no relu; atomic segment-sum accumulate ----
  {
    const int m  = lane & 15;
    const int kh = lane >> 4;
    int seg[8];
#pragma unroll
    for (int r = 0; r < 8; ++r) {
      const int edge = edgeBase + kh * 8 + r;   // C/D row = r + kh*8
      seg[r] = (edge < N) ? segid[edge] : -1;
    }
#pragma unroll 1
    for (int p = 0; p < 8; p += 4) {
      v8f acc[4] = {};
      gemm_pass<256, 8, 4>(wpk4, bufB, p, m, kh, lane, acc);
#pragma unroll
      for (int t = 0; t < 4; ++t) {
        const int ncol = (p + t) * 16 + m;
        const float bv = b4[ncol];
#pragma unroll
        for (int r = 0; r < 8; ++r) {
          if (seg[r] >= 0)
            atomicAdd(sums + seg[r] * FEAT + ncol, acc[t][r] + bv);
        }
      }
    }
  }
}

// ------------------------------- launcher ----------------------------------
extern "C" void kernel_launch(void* const* d_in, const int* in_sizes, int n_in,
                              void* d_out, int out_size, void* d_ws, size_t ws_size,
                              hipStream_t stream) {
  (void)n_in; (void)out_size; (void)ws_size;
  const float* g  = (const float*)d_in[0];
  const float* ns = (const float*)d_in[1];
  const float* nr = (const float*)d_in[2];
  const float* ef = (const float*)d_in[3];
  const int*   ei = (const int*)d_in[4];
  const float* W1 = (const float*)d_in[5];
  const float* b1 = (const float*)d_in[6];
  const float* W2 = (const float*)d_in[7];
  const float* b2 = (const float*)d_in[8];
  const float* W3 = (const float*)d_in[9];
  const float* b3 = (const float*)d_in[10];
  const float* W4 = (const float*)d_in[11];
  const float* b4 = (const float*)d_in[12];
  const int N = in_sizes[4];

  // workspace layout (bytes)
  char* ws = (char*)d_ws;
  _Float16* wpk1 = (_Float16*)(ws);                       // 512*256*2 = 262144
  _Float16* wpk2 = (_Float16*)(ws + 262144);              // 256*256*2 = 131072
  _Float16* wpk3 = (_Float16*)(ws + 393216);              // 256*256*2 = 131072
  _Float16* wpk4 = (_Float16*)(ws + 524288);              // 256*128*2 =  65536
  float*    sums   = (float*)(ws + 589824);               // 1024*128*4 = 524288
  float*    counts = (float*)(ws + 589824 + 524288);      // 1024*4     =   4096
  // (sums & counts contiguous: zero them together)

  pack_w_kernel<<<(512 * 256 + 255) / 256, 256, 0, stream>>>(W1, wpk1, 512, 256);
  pack_w_kernel<<<(256 * 256 + 255) / 256, 256, 0, stream>>>(W2, wpk2, 256, 256);
  pack_w_kernel<<<(256 * 256 + 255) / 256, 256, 0, stream>>>(W3, wpk3, 256, 256);
  pack_w_kernel<<<(256 * 128 + 255) / 256, 256, 0, stream>>>(W4, wpk4, 256, 128);

  const int zeroN = 1024 * 128 + 1024;
  zero_kernel<<<(zeroN + 255) / 256, 256, 0, stream>>>(sums, zeroN);

  const int nBlocks = (N + 127) / 128;
  const size_t ldsBytes = (size_t)8 * (16 * (DIN + ROWPAD) + 16 * (256 + ROWPAD)) * sizeof(_Float16);
  edgeblock_mlp_wmma<<<nBlocks, 256, ldsBytes, stream>>>(
      g, ns, nr, ef, ei, wpk1, wpk2, wpk3, wpk4, b1, b2, b3, b4, sums, N);

  count_kernel<<<(N + 255) / 256, 256, 0, stream>>>(ei, counts, N);

  const int outN = N * FEAT;
  finalize_kernel<<<(outN + 255) / 256, 256, 0, stream>>>(sums, counts, ei, (float*)d_out, N);
}
